// GraphAwareNodeModel_65292092833799
// MI455X (gfx1250) — matrix operand order repaired and checked by reading, not compiled, not run on hardware
//
#include <hip/hip_runtime.h>

// ---------------------------------------------------------------------------
// GraphAwareNodeModel for MI455X (gfx1250, wave32, WMMA 16x16x32 bf16)
//
// edge MLP:  h = LN(leaky(concat(x[src],ea,u) @ W1a + b1a)) * g1 + be1 @ W1b + b1b
// scatter-mean by dest -> agg
// node MLP:  out = LN(leaky(concat(x,agg) @ W2a + b2a)) * g2 + be2 @ W2b + b2b
//
// Weights are packed once into per-lane WMMA B-fragments and staged in LDS per
// block; each wave owns a 16-row tile and grid-strides over tiles.
// ---------------------------------------------------------------------------

typedef __bf16 bf16_t;
typedef __attribute__((ext_vector_type(16))) __bf16 v16bf;
typedef __attribute__((ext_vector_type(8)))  __bf16 v8bf;
typedef __attribute__((ext_vector_type(8)))  float  v8f;

#define NEG_SLOPE 0.01f
#define LN_EPS    1e-5f

#define IN_STRIDE 264   // bf16 elems per input-tile row (max 256 cols + pad; 132 dw = 4 banks mod 64)
#define MF_STRIDE 132   // f32  elems per mid-tile row   (overlays input tile)
#define MB_STRIDE 136   // bf16 elems per bf16 mid row   (68 dw = 4 banks mod 64)
#define WSCR      (16*IN_STRIDE*2 + 16*MB_STRIDE*2)   // 8448 + 4352 = 12800 B per wave
#define WAVES_PER_BLOCK 8
#define THREADS (32 * WAVES_PER_BLOCK)

#define FRAG_ELEMS 512          // one 32x16 bf16 fragment
#define WB_FRAGS   (4 * 8)      // second-layer GEMM: K=128 -> 4 ktiles x 8 ntiles

// ---- fp32 -> bf16 (round-to-nearest-even) ---------------------------------
static __device__ __forceinline__ unsigned bfbits(float x) {
  union { float f; unsigned u; } a; a.f = x;
  unsigned r = a.u + 0x7FFFu + ((a.u >> 16) & 1u);
  return r >> 16;
}
static __device__ __forceinline__ bf16_t f2bf(float x) {
  unsigned short h = (unsigned short)bfbits(x);
  bf16_t b; __builtin_memcpy(&b, &h, 2); return b;
}
static __device__ __forceinline__ unsigned pk2(float a, float b) {
  return bfbits(a) | (bfbits(b) << 16);
}

// ---- block-cooperative global->LDS copies ---------------------------------
static __device__ __forceinline__ void copy_g2l(bf16_t* __restrict__ d,
                                                const bf16_t* __restrict__ s,
                                                int elems) {
  const uint4* sp = (const uint4*)s;
  uint4* dp = (uint4*)d;
  int n = elems >> 3;
  for (int i = threadIdx.x; i < n; i += THREADS) dp[i] = sp[i];
}
static __device__ __forceinline__ void copy_vec(float* __restrict__ d,
                                                const float* __restrict__ s, int n) {
  for (int i = threadIdx.x; i < n; i += THREADS) d[i] = s[i];
}

// ---- 16xK @ Kx128 GEMM tile ----------------------------------------------
// A in LDS (bf16, row stride astride); B fragments in LDS, plane-split layout:
// fragment f occupies [f*512, f*512+512); plane p (p = e/8) at +p*256,
// lane chunk at +lane*8 -> ds_load_b128 with perfect 64-bank tiling.
template<int NKT>
static __device__ __forceinline__ void wmma_gemm(const bf16_t* __restrict__ ldsA,
                                                 int astride,
                                                 const bf16_t* __restrict__ ldsW,
                                                 int lane, v8f acc[8]) {
  const int half = lane >> 4;
  const int mrow = lane & 15;
#pragma unroll
  for (int kt = 0; kt < NKT; ++kt) {
    // 16-bit A layout: lane<16 holds K {0..7,16..23}, lane>=16 holds {8..15,24..31}
    const v8bf* ap = (const v8bf*)(ldsA + mrow * astride + kt * 32 + half * 8);
    v8bf a0 = ap[0];
    v8bf a1 = ap[2];   // +16 elements
    v16bf afrag;
#pragma unroll
    for (int e = 0; e < 8; ++e) { afrag[e] = a0[e]; afrag[8 + e] = a1[e]; }
#pragma unroll
    for (int nt = 0; nt < 8; ++nt) {
      const v8bf* bp = (const v8bf*)(ldsW + (((kt << 3) + nt) << 9) + (lane << 3));
      v8bf b0 = bp[0];
      v8bf b1 = bp[32];   // +256 elems (plane 1)
      v16bf bfrag;
#pragma unroll
      for (int e = 0; e < 8; ++e) { bfrag[e] = b0[e]; bfrag[8 + e] = b1[e]; }
      acc[nt] = __builtin_amdgcn_wmma_f32_16x16x32_bf16(
          false, afrag, false, bfrag, (short)0, acc[nt], false, false);
    }
  }
}

// ---- full MLP for one 16-row tile: GEMM1 + bias + leaky + LN + GEMM2 ------
// cvec (LDS): [biasA | g | be | biasB], 4 x 128 f32.  midf overlays inA.
template<int NKT1>
static __device__ __forceinline__ void mlp_tile(bf16_t* inA, bf16_t* midb,
                                                const bf16_t* __restrict__ wA,
                                                const bf16_t* __restrict__ wB,
                                                const float* __restrict__ cvec,
                                                int lane, v8f out[8]) {
  const int half = lane >> 4;
  const int ncol = lane & 15;
  float* midf = (float*)inA;   // overlay: input tile is dead after GEMM1

  v8f acc[8] = {};
  wmma_gemm<NKT1>(inA, IN_STRIDE, wA, lane, acc);

  // C/D layout: N = lane%16, M = r + 8*(lane/16).  bias + LeakyReLU -> LDS f32
#pragma unroll
  for (int nt = 0; nt < 8; ++nt) {
    float bb = cvec[nt * 16 + ncol];
#pragma unroll
    for (int r = 0; r < 8; ++r) {
      float v = acc[nt][r] + bb;
      v = (v >= 0.0f) ? v : NEG_SLOPE * v;
      midf[(r + 8 * half) * MF_STRIDE + nt * 16 + ncol] = v;
    }
  }
  asm volatile("s_wait_dscnt 0" ::: "memory");

  // LayerNorm over 128 cols: lane handles row lane%16, half of the columns
  const int row = lane & 15;
  const float* mr = midf + row * MF_STRIDE + half * 64;
  float s = 0.0f, s2 = 0.0f;
#pragma unroll
  for (int c = 0; c < 64; ++c) { float v = mr[c]; s += v; s2 += v * v; }
  s  += __shfl_xor(s, 16, 32);
  s2 += __shfl_xor(s2, 16, 32);
  float mu   = s * (1.0f / 128.0f);
  float var  = s2 * (1.0f / 128.0f) - mu * mu;
  float rinv = rsqrtf(var + LN_EPS);
#pragma unroll
  for (int c = 0; c < 64; ++c) {
    int col = half * 64 + c;
    float v = (mr[c] - mu) * rinv * cvec[128 + col] + cvec[256 + col];
    midb[row * MB_STRIDE + col] = f2bf(v);
  }
  asm volatile("s_wait_dscnt 0" ::: "memory");

#pragma unroll
  for (int nt = 0; nt < 8; ++nt) { v8f z = {}; out[nt] = z; }
  wmma_gemm<4>(midb, MB_STRIDE, wB, lane, out);
}

// ---------------------------------------------------------------------------
// Edge kernel: gather+concat -> MLP1 -> atomic scatter-add (sums, cnt)
// LDS: [W1a frags 57344B][W1b frags 32768B][cvec 2048B][8 x 12800B scratch]
// ---------------------------------------------------------------------------
#define E_WA_ELEMS (7 * 8 * FRAG_ELEMS)
#define E_SCR_OFF  (E_WA_ELEMS * 2 + WB_FRAGS * FRAG_ELEMS * 2 + 2048)
#define E_LDS      (E_SCR_OFF + WAVES_PER_BLOCK * WSCR)

__global__ __launch_bounds__(THREADS, 1)
void edge_kernel(const float* __restrict__ x, const int* __restrict__ src,
                 const int* __restrict__ dst, const float* __restrict__ ea,
                 const float* __restrict__ u,
                 const bf16_t* __restrict__ w1a, const float* __restrict__ b1a,
                 const float* __restrict__ g1,  const float* __restrict__ be1,
                 const bf16_t* __restrict__ w1b, const float* __restrict__ b1b,
                 float* __restrict__ sums, float* __restrict__ cnt,
                 int E, int ntiles) {
  extern __shared__ char smem[];
  bf16_t* wA   = (bf16_t*)smem;
  bf16_t* wB   = wA + E_WA_ELEMS;
  float*  cvec = (float*)(smem + E_WA_ELEMS * 2 + WB_FRAGS * FRAG_ELEMS * 2);

  copy_g2l(wA, w1a, E_WA_ELEMS);
  copy_g2l(wB, w1b, WB_FRAGS * FRAG_ELEMS);
  copy_vec(cvec +   0, b1a, 128);
  copy_vec(cvec + 128, g1,  128);
  copy_vec(cvec + 256, be1, 128);
  copy_vec(cvec + 384, b1b, 128);
  __syncthreads();

  const int lane = threadIdx.x & 31;
  const int wave = threadIdx.x >> 5;
  const int half = lane >> 4;
  const int ncol = lane & 15;

  char* scr = smem + E_SCR_OFF + wave * WSCR;
  bf16_t* inA  = (bf16_t*)scr;
  bf16_t* midb = (bf16_t*)(scr + 16 * IN_STRIDE * 2);

  const bf16_t ub = f2bf(u[lane]);   // DG == 32, one per lane

  for (int tile = blockIdx.x * WAVES_PER_BLOCK + wave; tile < ntiles;
       tile += gridDim.x * WAVES_PER_BLOCK) {
    const int tb = tile * 16;
    int erl = tb + (lane & 15); if (erl >= E) erl = E - 1;
    const int srcv = src[erl];
    const int dstv = dst[erl];

    // Stage 16 x 224 input tile (bf16): x[src] | edge_attr | u
#pragma unroll 1
    for (int r = 0; r < 16; ++r) {
      int srow = __shfl(srcv, r, 32);
      float4 xv = *((const float4*)(x + (size_t)srow * 128) + lane);
      *(uint2*)(inA + r * IN_STRIDE + lane * 4) = make_uint2(pk2(xv.x, xv.y), pk2(xv.z, xv.w));
      int erow = tb + r; if (erow >= E) erow = E - 1;
      float2 ev = *((const float2*)(ea + (size_t)erow * 64) + lane);
      *(unsigned*)(inA + r * IN_STRIDE + 128 + lane * 2) = pk2(ev.x, ev.y);
      inA[r * IN_STRIDE + 192 + lane] = ub;
    }
    asm volatile("s_wait_dscnt 0" ::: "memory");

    v8f h[8];
    mlp_tile<7>(inA, midb, wA, wB, cvec, lane, h);

#pragma unroll
    for (int nt = 0; nt < 8; ++nt) {
      float bb = cvec[384 + nt * 16 + ncol];
#pragma unroll
      for (int r = 0; r < 8; ++r) {
        int M = r + 8 * half;
        int e = tb + M;
        int dm = __shfl(dstv, M, 32);
        if (e < E)
          atomicAdd(&sums[(size_t)dm * 128 + nt * 16 + ncol], h[nt][r] + bb);
      }
    }
    if (lane < 16 && (tb + lane) < E) atomicAdd(&cnt[dstv], 1.0f);
  }
}

// ---------------------------------------------------------------------------
// Node kernel: concat(x, sums/max(cnt,1)) -> MLP2 -> out
// LDS: [W2a frags 65536B][W2b frags 32768B][cvec 2048B][8 x 12800B scratch]
// ---------------------------------------------------------------------------
#define N_WA_ELEMS (8 * 8 * FRAG_ELEMS)
#define N_SCR_OFF  (N_WA_ELEMS * 2 + WB_FRAGS * FRAG_ELEMS * 2 + 2048)
#define N_LDS      (N_SCR_OFF + WAVES_PER_BLOCK * WSCR)

__global__ __launch_bounds__(THREADS, 1)
void node_kernel(const float* __restrict__ x, const float* __restrict__ sums,
                 const float* __restrict__ cnt,
                 const bf16_t* __restrict__ w2a, const float* __restrict__ b2a,
                 const float* __restrict__ g2,  const float* __restrict__ be2,
                 const bf16_t* __restrict__ w2b, const float* __restrict__ b2b,
                 float* __restrict__ out, int N, int ntiles) {
  extern __shared__ char smem[];
  bf16_t* wA   = (bf16_t*)smem;
  bf16_t* wB   = wA + N_WA_ELEMS;
  float*  cvec = (float*)(smem + N_WA_ELEMS * 2 + WB_FRAGS * FRAG_ELEMS * 2);

  copy_g2l(wA, w2a, N_WA_ELEMS);
  copy_g2l(wB, w2b, WB_FRAGS * FRAG_ELEMS);
  copy_vec(cvec +   0, b2a, 128);
  copy_vec(cvec + 128, g2,  128);
  copy_vec(cvec + 256, be2, 128);
  copy_vec(cvec + 384, b2b, 128);
  __syncthreads();

  const int lane = threadIdx.x & 31;
  const int wave = threadIdx.x >> 5;
  const int half = lane >> 4;
  const int ncol = lane & 15;

  char* scr = smem + N_SCR_OFF + wave * WSCR;
  bf16_t* inA  = (bf16_t*)scr;
  bf16_t* midb = (bf16_t*)(scr + 16 * IN_STRIDE * 2);

  for (int tile = blockIdx.x * WAVES_PER_BLOCK + wave; tile < ntiles;
       tile += gridDim.x * WAVES_PER_BLOCK) {
    const int tb = tile * 16;
    int nrl = tb + (lane & 15); if (nrl >= N) nrl = N - 1;
    float civ = 1.0f / fmaxf(cnt[nrl], 1.0f);

    // Stage 16 x 256 input tile (bf16): x | agg
#pragma unroll 1
    for (int r = 0; r < 16; ++r) {
      int nrow = tb + r; if (nrow >= N) nrow = N - 1;
      float4 xv = *((const float4*)(x + (size_t)nrow * 128) + lane);
      *(uint2*)(inA + r * IN_STRIDE + lane * 4) = make_uint2(pk2(xv.x, xv.y), pk2(xv.z, xv.w));
      float ci = __shfl(civ, r, 32);
      float4 sv = *((const float4*)(sums + (size_t)nrow * 128) + lane);
      *(uint2*)(inA + r * IN_STRIDE + 128 + lane * 4) =
          make_uint2(pk2(sv.x * ci, sv.y * ci), pk2(sv.z * ci, sv.w * ci));
    }
    asm volatile("s_wait_dscnt 0" ::: "memory");

    v8f h[8];
    mlp_tile<8>(inA, midb, wA, wB, cvec, lane, h);

#pragma unroll
    for (int nt = 0; nt < 8; ++nt) {
      float bb = cvec[384 + nt * 16 + ncol];
#pragma unroll
      for (int r = 0; r < 8; ++r) {
        int M = r + 8 * half;
        int n = tb + M;
        if (n < N) out[(size_t)n * 128 + nt * 16 + ncol] = h[nt][r] + bb;
      }
    }
  }
}

// ---------------------------------------------------------------------------
// Weight repack: f32 row-major [K,128] -> bf16 WMMA B-fragments, plane-split:
// frag (kt,nt), element (lane,e):  W[kt*32 + (e/8)*16 + (lane/16)*8 + e%8][nt*16 + lane%16]
// stored at frag*512 + (e/8)*256 + lane*8 + e%8   (conflict-free ds_load_b128)
// ---------------------------------------------------------------------------
__global__ void pack_w(const float* __restrict__ W, bf16_t* __restrict__ outw, int nkt) {
  int t = blockIdx.x * 256 + threadIdx.x;
  int total = nkt * 8 * FRAG_ELEMS;
  if (t >= total) return;
  int frag  = t >> 9, idx = t & 511;
  int plane = idx >> 8;
  int lane  = (idx >> 3) & 31;
  int e     = (plane << 3) + (idx & 7);
  int kt = frag >> 3, nt = frag & 7;
  int k = kt * 32 + ((e >> 3) << 4) + ((lane >> 4) << 3) + (e & 7);
  int n = nt * 16 + (lane & 15);
  outw[t] = f2bf(W[(size_t)k * 128 + n]);
}

__global__ void zero_kernel(float* __restrict__ p, size_t n) {
  size_t i = (size_t)blockIdx.x * blockDim.x + threadIdx.x;
  size_t stride = (size_t)gridDim.x * blockDim.x;
  for (; i < n; i += stride) p[i] = 0.0f;
}

// ---------------------------------------------------------------------------
extern "C" void kernel_launch(void* const* d_in, const int* in_sizes, int n_in,
                              void* d_out, int out_size, void* d_ws, size_t ws_size,
                              hipStream_t stream) {
  const float* x   = (const float*)d_in[0];
  const int*   ei  = (const int*)  d_in[1];
  const float* ea  = (const float*)d_in[2];
  const float* u   = (const float*)d_in[3];
  // d_in[4] = batch (unused)
  const float* W1a = (const float*)d_in[5];
  const float* b1a = (const float*)d_in[6];
  const float* g1  = (const float*)d_in[7];
  const float* be1 = (const float*)d_in[8];
  const float* W1b = (const float*)d_in[9];
  const float* b1b = (const float*)d_in[10];
  const float* W2a = (const float*)d_in[11];
  const float* b2a = (const float*)d_in[12];
  const float* g2  = (const float*)d_in[13];
  const float* be2 = (const float*)d_in[14];
  const float* W2b = (const float*)d_in[15];
  const float* b2b = (const float*)d_in[16];

  const int N = in_sizes[0] / 128;
  const int E = in_sizes[2] / 64;
  const int* src = ei;
  const int* dst = ei + E;

  // workspace: sums[N*128] | cnt[N] | packed bf16 weight fragments
  float* sums = (float*)d_ws;
  float* cnt  = sums + (size_t)N * 128;
  size_t woff = (((size_t)N * 129 * sizeof(float)) + 255) & ~(size_t)255;
  bf16_t* w1a = (bf16_t*)((char*)d_ws + woff);
  bf16_t* w1b = w1a + 7 * 8 * FRAG_ELEMS;
  bf16_t* w2a = w1b + 4 * 8 * FRAG_ELEMS;
  bf16_t* w2b = w2a + 8 * 8 * FRAG_ELEMS;

  zero_kernel<<<2048, 256, 0, stream>>>(sums, (size_t)N * 129);
  pack_w<<<(7 * 8 * FRAG_ELEMS + 255) / 256, 256, 0, stream>>>(W1a, w1a, 7);
  pack_w<<<(4 * 8 * FRAG_ELEMS + 255) / 256, 256, 0, stream>>>(W1b, w1b, 4);
  pack_w<<<(8 * 8 * FRAG_ELEMS + 255) / 256, 256, 0, stream>>>(W2a, w2a, 8);
  pack_w<<<(4 * 8 * FRAG_ELEMS + 255) / 256, 256, 0, stream>>>(W2b, w2b, 4);

  (void)hipFuncSetAttribute((const void*)edge_kernel,
                            hipFuncAttributeMaxDynamicSharedMemorySize, E_LDS);
  (void)hipFuncSetAttribute((const void*)node_kernel,
                            hipFuncAttributeMaxDynamicSharedMemorySize, N_LDS);

  const int etiles  = (E + 15) / 16;
  int eblocks = (etiles + WAVES_PER_BLOCK - 1) / WAVES_PER_BLOCK;
  if (eblocks > 2048) eblocks = 2048;
  edge_kernel<<<eblocks, THREADS, E_LDS, stream>>>(
      x, src, dst, ea, u, w1a, b1a, g1, be1, w1b, b1b, sums, cnt, E, etiles);

  const int ntiles  = (N + 15) / 16;
  int nblocks = (ntiles + WAVES_PER_BLOCK - 1) / WAVES_PER_BLOCK;
  if (nblocks > 2048) nblocks = 2048;
  node_kernel<<<nblocks, THREADS, N_LDS, stream>>>(
      x, sums, cnt, w2a, b2a, g2, be2, w2b, b2b, (float*)d_out, N, ntiles);
}
